// SRUppAttention_42477226557452
// MI455X (gfx1250) — compile-verified, matrix operand order checked by reading, not run on hardware
//
#include <hip/hip_runtime.h>
#include <hip/hip_bf16.h>

typedef __attribute__((ext_vector_type(16))) _Float16 v16h;
typedef __attribute__((ext_vector_type(8)))  _Float16 v8h;
typedef __attribute__((ext_vector_type(4)))  _Float16 v4h;
typedef __attribute__((ext_vector_type(8)))  float    v8f;
typedef __attribute__((ext_vector_type(4)))  float    v4f;

#define WMMA_F16(A, B, C) \
    __builtin_amdgcn_wmma_f32_16x16x32_f16(false, (A), false, (B), (short)0, (C), false, false)

// ---------------------------------------------------------------------------
// LDS byte offset of a __shared__ object: generic LDS addresses map to LDS via
// addr[31:0] (ISA 10.2 aperture calc).
// ---------------------------------------------------------------------------
static __device__ __forceinline__ unsigned lds_off(const void* p) {
    return (unsigned)(uintptr_t)p;
}

// global -> LDS async copy, 16B per lane.  Tracked by ASYNCcnt.
static __device__ __forceinline__ void async_ld_b128(unsigned lds_byte, const void* gaddr) {
    asm volatile("global_load_async_to_lds_b128 %0, %1, off"
                 :: "v"(lds_byte), "v"(gaddr) : "memory");
}
static __device__ __forceinline__ void wait_async(int n) {
    if (n) asm volatile("s_wait_asynccnt 2" ::: "memory");
    else   asm volatile("s_wait_asynccnt 0" ::: "memory");
}

// ---------------------------------------------------------------------------
// DPP 16-lane butterfly reductions (DPP row = 16 lanes, wave32).
// ---------------------------------------------------------------------------
template <int CTRL>
static __device__ __forceinline__ float dpp_xfer(float x) {
    int r = __builtin_amdgcn_update_dpp(0, __builtin_bit_cast(int, x), CTRL, 0xf, 0xf, true);
    return __builtin_bit_cast(float, r);
}
static __device__ __forceinline__ float row16_max(float v) {
    v = fmaxf(v, dpp_xfer<0xB1>(v));    // quad_perm [1,0,3,2]
    v = fmaxf(v, dpp_xfer<0x4E>(v));    // quad_perm [2,3,0,1]
    v = fmaxf(v, dpp_xfer<0x141>(v));   // row_half_mirror
    v = fmaxf(v, dpp_xfer<0x140>(v));   // row_mirror
    return v;
}
static __device__ __forceinline__ float row16_sum(float v) {
    v += dpp_xfer<0xB1>(v);
    v += dpp_xfer<0x4E>(v);
    v += dpp_xfer<0x141>(v);
    v += dpp_xfer<0x140>(v);
    return v;
}

// ---------------------------------------------------------------------------
// Fragment loader for V_WMMA_F32_16X16X32_F16, K-major storage (global or LDS).
// ---------------------------------------------------------------------------
static __device__ __forceinline__ v16h ld_frag_kmaj(const _Float16* base, int ld, int lane) {
    const int r  = lane & 15;
    const int hi = (lane >> 4) << 3;  // 0 or 8
    const _Float16* p = base + (size_t)r * ld + hi;
    union { v16h v; v8h h[2]; } u;
    u.h[0] = *reinterpret_cast<const v8h*>(p);       // K = hi .. hi+7
    u.h[1] = *reinterpret_cast<const v8h*>(p + 16);  // K = 16+hi .. 16+hi+7
    return u.v;
}

// ---------------------------------------------------------------------------
// f32 -> f16 conversion, 4 elements/thread
// ---------------------------------------------------------------------------
__global__ void cvt_f32_to_f16_v4(const float* __restrict__ in, _Float16* __restrict__ out, int n4) {
    int i = blockIdx.x * blockDim.x + threadIdx.x;
    if (i < n4) {
        v4f x = reinterpret_cast<const v4f*>(in)[i];
        v4h y;
#pragma unroll
        for (int j = 0; j < 4; ++j) y[j] = (_Float16)x[j];
        reinterpret_cast<v4h*>(out)[i] = y;
    }
}

// ---------------------------------------------------------------------------
// C[M,N] = A[M,K] @ W[N,K]^T      (f16 in, f32 accumulate)
// 256 threads = 8 waves; block tile 128(M) x 64(N); K staged in 32-wide slabs
// through double-buffered LDS via async DMA (B slab loaded once per block).
// MODE 0: f32 out.  MODE 1: f32 + f16 out.  MODE 2: kv-split f16 out:
//   n<512 -> Kh[m,n] ; n>=512 -> transposed Vt[(b*8+h)*64+d][l] (keys-major).
// ---------------------------------------------------------------------------
template <int MODE, int N, int K>
__global__ __launch_bounds__(256)
void wmma_gemm_wt(const _Float16* __restrict__ A, const _Float16* __restrict__ W,
                  float* __restrict__ Cf, _Float16* __restrict__ Ch,
                  _Float16* __restrict__ Vt, int M) {
    __shared__ _Float16 sA[2][128 * 32];
    __shared__ _Float16 sB[2][64 * 32];

    const int tid  = threadIdx.x;
    const int lane = tid & 31;
    const int wave = tid >> 5;
    const int m0 = blockIdx.x * 128;
    const int n0 = blockIdx.y * 64;

    // 16B chunk map: A slab 128x32 = 512 chunks (2/thread); B slab 64x32 = 256.
    const int rowA = tid >> 2;           // 0..63
    const int qA   = (tid & 3) * 8;
    const _Float16* gA0 = A + (size_t)(m0 + rowA) * K + qA;
    const _Float16* gA1 = A + (size_t)(m0 + rowA + 64) * K + qA;
    const _Float16* gB  = W + (size_t)(n0 + rowA) * K + qA;

    constexpr int nk = K >> 5;

    auto issue = [&](int t, int buf) {
        const int k = t << 5;
        async_ld_b128(lds_off(&sA[buf][rowA * 32 + qA]),        gA0 + k);
        async_ld_b128(lds_off(&sA[buf][(rowA + 64) * 32 + qA]), gA1 + k);
        async_ld_b128(lds_off(&sB[buf][rowA * 32 + qA]),        gB + k);
    };

    v8f acc[4] = {v8f{0}, v8f{0}, v8f{0}, v8f{0}};

    issue(0, 0);
    for (int t = 0; t < nk; ++t) {
        const int cur = t & 1;
        if (t + 1 < nk) { issue(t + 1, cur ^ 1); wait_async(1); }
        else            { wait_async(0); }
        __syncthreads();                         // all waves' slab-t DMA done

        v16h a  = ld_frag_kmaj(&sA[cur][(wave * 16) * 32], 32, lane);
        v16h b0 = ld_frag_kmaj(&sB[cur][0 * 16 * 32], 32, lane);
        v16h b1 = ld_frag_kmaj(&sB[cur][1 * 16 * 32], 32, lane);
        v16h b2 = ld_frag_kmaj(&sB[cur][2 * 16 * 32], 32, lane);
        v16h b3 = ld_frag_kmaj(&sB[cur][3 * 16 * 32], 32, lane);
        acc[0] = WMMA_F16(a, b0, acc[0]);
        acc[1] = WMMA_F16(a, b1, acc[1]);
        acc[2] = WMMA_F16(a, b2, acc[2]);
        acc[3] = WMMA_F16(a, b3, acc[3]);

        __syncthreads();                         // reads done before overwrite
    }

    const int mh = (lane >> 4) << 3;
    const int cn = lane & 15;
    const int mw = m0 + wave * 16;
#pragma unroll
    for (int r = 0; r < 8; ++r) {
        const int m = mw + mh + r;
#pragma unroll
        for (int j = 0; j < 4; ++j) {
            const float v = acc[j][r];
            if (MODE == 0 || MODE == 1) Cf[(size_t)m * N + n0 + cn + j * 16] = v;
            if (MODE == 1) Ch[(size_t)m * N + n0 + cn + j * 16] = (_Float16)v;
            if (MODE == 2) {
                const int n = n0 + j * 16 + cn;
                if (n < 512) {
                    Ch[(size_t)m * 512 + n] = (_Float16)v;
                } else {
                    const int p = n - 512;            // h*64 + d
                    const int bb = m & 3, l = m >> 2; // m = l*4 + b
                    Vt[((size_t)(bb * 8 + (p >> 6)) * 64 + (p & 63)) * 2048 + l] = (_Float16)v;
                }
            }
        }
    }
}

// ---------------------------------------------------------------------------
// Flash attention.  grid = (L/64, B*H); 128 threads = 4 waves, each wave owns
// 16 query rows.  K tiles async-staged (double-buffered, shared by 4 waves);
// V read as pre-transposed vT[d][key] fragments straight from global/L2;
// DPP row reductions for online softmax; P relayout C->A via 1KB/wave LDS.
// ---------------------------------------------------------------------------
__global__ __launch_bounds__(128)
void wmma_flash_attn(const _Float16* __restrict__ zh,    // Q source [L*B, 512]
                     const _Float16* __restrict__ kh,    // K        [L*B, 512]
                     const _Float16* __restrict__ vT,    // V^T      [32][64][L]
                     const float* __restrict__ amask,    // [L, L]
                     const unsigned char* __restrict__ pad, // [L, B] bool
                     float* __restrict__ outf) {         // [L*B, 512]
    constexpr int L = 2048, B = 4;
    __shared__ _Float16 sK[2][32 * 64];
    __shared__ _Float16 pbuf[4][16 * 32];

    const int tid  = threadIdx.x;
    const int lane = tid & 31;
    const int wave = tid >> 5;
    const int bh = blockIdx.y;
    const int b = bh >> 3;
    const int h = bh & 7;
    const int q0 = blockIdx.x * 64 + wave * 16;

    constexpr int ldq = B * 512;   // 2048
    const _Float16* qbase = zh + (size_t)b * 512 + h * 64;
    const _Float16* kbase = kh + (size_t)b * 512 + h * 64;
    const _Float16* vbase = vT + (size_t)bh * 64 * 2048;

    // K tile = 32 keys x 64 d = 256 16B-chunks; 128 threads -> 2 each.
    const int rowT = tid >> 3;          // 0..15
    const int qT   = (tid & 7) * 8;
    const _Float16* gk0 = kbase + (size_t)rowT * ldq + qT;
    const _Float16* gk1 = kbase + (size_t)(rowT + 16) * ldq + qT;
    auto issue = [&](int kt, int buf) {
        async_ld_b128(lds_off(&sK[buf][rowT * 64 + qT]),        gk0 + (size_t)kt * ldq);
        async_ld_b128(lds_off(&sK[buf][(rowT + 16) * 64 + qT]), gk1 + (size_t)kt * ldq);
    };

    _Float16* pb = &pbuf[wave][0];
    const int mh = (lane >> 4) << 3;
    const int cn = lane & 15;
    const float scaling = 0.125f;   // 64^-0.5

    // Q fragments (reused for every key tile)
    v16h qf0 = ld_frag_kmaj(qbase + (size_t)q0 * ldq + 0,  ldq, lane);
    v16h qf1 = ld_frag_kmaj(qbase + (size_t)q0 * ldq + 32, ldq, lane);

    // per-tile gather bases with constant strides (fold into imm offsets)
    const float* am = amask + (size_t)(q0 + mh) * L + cn;   // + r*L + kt(+16)
    const unsigned char* pp = pad + b;                      // + (kt+cn(+16))*B

    v8f o[4] = {v8f{0}, v8f{0}, v8f{0}, v8f{0}};
    float mrow[8], lrow[8];
#pragma unroll
    for (int r = 0; r < 8; ++r) { mrow[r] = -1e30f; lrow[r] = 0.0f; }

    constexpr int nt = L >> 5;
    issue(0, 0);
    for (int t = 0; t < nt; ++t) {
        const int cur = t & 1;
        const int kt = t << 5;
        if (t + 1 < nt) { issue(kt + 32, cur ^ 1); wait_async(1); }
        else            { wait_async(0); }
        __syncthreads();                          // K tile t in LDS

        // ---- S = Q K^T (16 x 32) -> two 16x16 C fragments ----
        v16h b00 = ld_frag_kmaj(&sK[cur][0 * 64 + 0],   64, lane);
        v16h b01 = ld_frag_kmaj(&sK[cur][0 * 64 + 32],  64, lane);
        v16h b10 = ld_frag_kmaj(&sK[cur][16 * 64 + 0],  64, lane);
        v16h b11 = ld_frag_kmaj(&sK[cur][16 * 64 + 32], 64, lane);
        v8f s0 = {}; v8f s1 = {};
        s0 = WMMA_F16(qf0, b00, s0);
        s0 = WMMA_F16(qf1, b01, s0);
        s1 = WMMA_F16(qf0, b10, s1);
        s1 = WMMA_F16(qf1, b11, s1);

        // ---- scale + additive mask + key-padding mask ----
        const unsigned char p0 = pp[(kt + cn) * B];
        const unsigned char p1 = pp[(kt + 16 + cn) * B];
        const float* amk = am + kt;
        float smax[8];
#pragma unroll
        for (int r = 0; r < 8; ++r) {
            float x0 = s0[r] * scaling + amk[r * L];
            float x1 = s1[r] * scaling + amk[r * L + 16];
            if (p0) x0 = -1e30f;
            if (p1) x1 = -1e30f;
            s0[r] = x0; s1[r] = x1;
            smax[r] = row16_max(fmaxf(x0, x1));
        }

        // ---- online softmax; stage P (f16) in LDS [row][key] ----
#pragma unroll
        for (int r = 0; r < 8; ++r) {
            const float mnew = fmaxf(mrow[r], smax[r]);
            const float corr = __expf(mrow[r] - mnew);
            const float e0 = __expf(s0[r] - mnew);
            const float e1 = __expf(s1[r] - mnew);
            lrow[r] = lrow[r] * corr + row16_sum(e0 + e1);
            mrow[r] = mnew;
#pragma unroll
            for (int j = 0; j < 4; ++j) o[j][r] *= corr;
            pb[(mh + r) * 32 + cn]      = (_Float16)e0;
            pb[(mh + r) * 32 + 16 + cn] = (_Float16)e1;
        }

        // ---- O += P (16x32) · V (32x64) ; V from transposed global ----
        v16h pf = ld_frag_kmaj(pb, 32, lane);
#pragma unroll
        for (int j = 0; j < 4; ++j) {
            v16h vb = ld_frag_kmaj(vbase + (size_t)(j * 16) * 2048 + kt, 2048, lane);
            o[j] = WMMA_F16(pf, vb, o[j]);
        }

        __syncthreads();                          // K reads done before overwrite
    }

    // ---- normalize and scatter back to [L, B, PROJ] layout (f32) ----
#pragma unroll
    for (int r = 0; r < 8; ++r) {
        const float inv = 1.0f / lrow[r];
        const int l = q0 + mh + r;
        const size_t base = ((size_t)l * B + b) * 512 + h * 64 + cn;
#pragma unroll
        for (int j = 0; j < 4; ++j) outf[base + j * 16] = o[j][r] * inv;
    }
}

// ---------------------------------------------------------------------------
// out_h = LayerNorm(attn*alpha + z) * gamma + beta  (one block per row, 512 cols)
// ---------------------------------------------------------------------------
__global__ __launch_bounds__(256)
void rezero_layernorm(const float* __restrict__ attn, const float* __restrict__ z,
                      const float* __restrict__ alpha, const float* __restrict__ gamma,
                      const float* __restrict__ beta, _Float16* __restrict__ outh) {
    const int row = blockIdx.x;
    const int t = threadIdx.x;
    const float a = alpha[0];
    const size_t base = (size_t)row * 512;

    const float x0 = attn[base + t]       * a + z[base + t];
    const float x1 = attn[base + 256 + t] * a + z[base + 256 + t];

    __shared__ float red[256];
    red[t] = x0 + x1;
    __syncthreads();
#pragma unroll
    for (int off = 128; off > 0; off >>= 1) {
        if (t < off) red[t] += red[t + off];
        __syncthreads();
    }
    const float mu = red[0] * (1.0f / 512.0f);
    __syncthreads();

    const float d0 = x0 - mu, d1 = x1 - mu;
    red[t] = d0 * d0 + d1 * d1;
    __syncthreads();
#pragma unroll
    for (int off = 128; off > 0; off >>= 1) {
        if (t < off) red[t] += red[t + off];
        __syncthreads();
    }
    const float rinv = rsqrtf(red[0] * (1.0f / 512.0f) + 1e-5f);

    outh[base + t]       = (_Float16)(d0 * rinv * gamma[t]       + beta[t]);
    outh[base + 256 + t] = (_Float16)(d1 * rinv * gamma[t + 256] + beta[t + 256]);
}

// ---------------------------------------------------------------------------
// Launcher
// ---------------------------------------------------------------------------
extern "C" void kernel_launch(void* const* d_in, const int* in_sizes, int n_in,
                              void* d_out, int out_size, void* d_ws, size_t ws_size,
                              hipStream_t stream) {
    (void)in_sizes; (void)n_in; (void)out_size; (void)ws_size;

    constexpr int L = 2048, B = 4, IN_F = 2048, PROJ = 512, OUT_F = 2048;
    constexpr int M = L * B;  // 8192

    const float* input            = (const float*)d_in[0];
    const unsigned char* mask_pad = (const unsigned char*)d_in[1];  // bool [L,B]
    const float* amask            = (const float*)d_in[2];          // [L,L]
    const float* W1               = (const float*)d_in[3];          // [PROJ, IN_F]
    const float* W2               = (const float*)d_in[4];          // [2*PROJ, PROJ]
    const float* W3               = (const float*)d_in[5];          // [OUT_F, PROJ]
    const float* alpha            = (const float*)d_in[6];
    const float* gamma            = (const float*)d_in[7];
    const float* beta             = (const float*)d_in[8];

    char* ws = (char*)d_ws;
    _Float16* in_h = (_Float16*)ws; ws += (size_t)M * IN_F * 2;        // 32 MB
    _Float16* w1h  = (_Float16*)ws; ws += (size_t)PROJ * IN_F * 2;     //  2 MB
    _Float16* w2h  = (_Float16*)ws; ws += (size_t)2 * PROJ * PROJ * 2; //  1 MB
    _Float16* w3h  = (_Float16*)ws; ws += (size_t)OUT_F * PROJ * 2;    //  2 MB
    float*    zf   = (float*)ws;    ws += (size_t)M * PROJ * 4;        // 16 MB
    _Float16* zh   = (_Float16*)ws; ws += (size_t)M * PROJ * 2;        //  8 MB
    _Float16* kh   = (_Float16*)ws; ws += (size_t)M * PROJ * 2;        //  8 MB
    _Float16* vTh  = (_Float16*)ws; ws += (size_t)M * PROJ * 2;        //  8 MB
    float*    attnf = (float*)ws;   ws += (size_t)M * PROJ * 4;        // 16 MB
    _Float16* normh = (_Float16*)ws;                                   //  8 MB

    // 0) f32 -> f16 conversions (4 elems/thread)
    cvt_f32_to_f16_v4<<<(M * IN_F / 4) / 256, 256, 0, stream>>>(input, in_h, M * IN_F / 4);
    cvt_f32_to_f16_v4<<<(PROJ * IN_F / 4) / 256, 256, 0, stream>>>(W1, w1h, PROJ * IN_F / 4);
    cvt_f32_to_f16_v4<<<(2 * PROJ * PROJ / 4) / 256, 256, 0, stream>>>(W2, w2h, 2 * PROJ * PROJ / 4);
    cvt_f32_to_f16_v4<<<(OUT_F * PROJ / 4) / 256, 256, 0, stream>>>(W3, w3h, OUT_F * PROJ / 4);

    // 1) z = input @ W1^T   (f32 residual + f16 operand copy)
    {
        dim3 grid(M / 128, PROJ / 64);
        wmma_gemm_wt<1, PROJ, IN_F><<<grid, 256, 0, stream>>>(in_h, w1h, zf, zh, nullptr, M);
    }
    // 2) kv = z @ W2^T   (K half normal, V half transposed per-head)
    {
        dim3 grid(M / 128, (2 * PROJ) / 64);
        wmma_gemm_wt<2, 2 * PROJ, PROJ><<<grid, 256, 0, stream>>>(zh, w2h, nullptr, kh, vTh, M);
    }
    // 3) flash attention over 32 batch-heads
    {
        dim3 grid(L / 64, B * 8);
        wmma_flash_attn<<<grid, 128, 0, stream>>>(zh, kh, vTh, amask, mask_pad, attnf);
    }
    // 4) rezero residual + post layer norm -> f16
    rezero_layernorm<<<M, 256, 0, stream>>>(attnf, zf, alpha, gamma, beta, normh);

    // 5) out = norm @ W3^T  (f32 into d_out)
    {
        dim3 grid(M / 128, OUT_F / 64);
        wmma_gemm_wt<0, OUT_F, PROJ><<<grid, 256, 0, stream>>>(normh, w3h, (float*)d_out,
                                                               nullptr, nullptr, M);
    }
}
